// HeteroGNN_44994077393231
// MI455X (gfx1250) — compile-verified, matrix operand order checked by reading, not run on hardware
//
#include <hip/hip_runtime.h>
#include <hip/hip_bf16.h>

typedef __attribute__((ext_vector_type(16))) _Float16 v16h;
typedef __attribute__((ext_vector_type(8)))  _Float16 v8h;
typedef __attribute__((ext_vector_type(8)))  float    v8f;

#define NU_  100000
#define NM_  100000
#define H_   128
#define DU_  32
#define DE_  16
#define EMP_ 1000000
#define ECL_ 500000

// Assemble a v16h WMMA fragment from two contiguous 16B runs in LDS.
// Per CDNA5 ISA 16-bit 16x32 layout: elements 0..7 hold K = khalf*8 + e,
// elements 8..15 hold K = 16 + khalf*8 + (e-8)  -> two aligned v8h loads.
union FragU { v16h v; v8h h[2]; };
__device__ __forceinline__ v16h frag16(const _Float16* __restrict__ p, int khalf) {
    FragU u;
    u.h[0] = *(const v8h*)(p + khalf * 8);
    u.h[1] = *(const v8h*)(p + 16 + khalf * 8);
    return u.v;
}

// -------------------------------------------------------------------------
// Edge-count kernel: cnt[dst] += 1 per edge
// -------------------------------------------------------------------------
__global__ __launch_bounds__(256)
void count_kernel(const int* __restrict__ ei_dst, float* __restrict__ cnt, int E) {
    int e = blockIdx.x * blockDim.x + threadIdx.x;
    if (e < E) atomicAdd(&cnt[ei_dst[e]], 1.0f);
}

// cnt -> 1/max(cnt,1), computed once (keeps the fp32 div chain out of GEMMs)
__global__ __launch_bounds__(256)
void invcnt_kernel(float* __restrict__ cnt, int n) {
    int i = blockIdx.x * blockDim.x + threadIdx.x;
    if (i < n) cnt[i] = 1.0f / fmaxf(cnt[i], 1.0f);
}

// -------------------------------------------------------------------------
// Scatter-add: agg[ei_dst[e]] += src[ei_src[e]]  (32 lanes/edge, float4/lane)
// -------------------------------------------------------------------------
__global__ __launch_bounds__(256)
void scatter_add_kernel(const float* __restrict__ src,
                        const int* __restrict__ ei_src,
                        const int* __restrict__ ei_dst,
                        float* __restrict__ agg, int E) {
    int idx = blockIdx.x * blockDim.x + threadIdx.x;
    int e = idx >> 5;
    if (e >= E) return;
    int c = (idx & 31) * 4;
    int s = ei_src[e], d = ei_dst[e];
    float4 v = *(const float4*)&src[(size_t)s * H_ + c];
    float* dst = &agg[(size_t)d * H_ + c];
    atomicAdd(dst + 0, v.x);
    atomicAdd(dst + 1, v.y);
    atomicAdd(dst + 2, v.z);
    atomicAdd(dst + 3, v.w);
}

// -------------------------------------------------------------------------
// One GEMM pass: acc += scale(X)@W for a 16x128 block tile.
// As: 16x32 f16 (row-major, lane K-runs contiguous).
// Bt: 128x32 f16 (B transposed, lane K-runs contiguous, 16B aligned).
// -------------------------------------------------------------------------
template<int K>
__device__ __forceinline__ v8f gemm_pass(const float* __restrict__ X,
                                         const float* __restrict__ W,
                                         float scale, int arow, int N,
                                         _Float16 (* __restrict__ As)[32],
                                         _Float16 (* __restrict__ Bt)[32],
                                         int ar, int ak, int bk, int bc,
                                         int ln, int khalf, int n0, v8f acc) {
    #pragma unroll
    for (int k0 = 0; k0 < K; k0 += 32) {
        __syncthreads();
        {   // stage A: 2 elems/thread, pre-scaled (mean) f32 -> f16
            float2 v = make_float2(0.0f, 0.0f);
            if (arow < N) v = *(const float2*)&X[(size_t)arow * K + k0 + ak];
            As[ar][ak]     = (_Float16)(v.x * scale);
            As[ar][ak + 1] = (_Float16)(v.y * scale);
        }
        {   // stage B transposed: 16 elems/thread
            const float* wsrc = &W[(size_t)(k0 + bk) * H_ + bc];
            #pragma unroll
            for (int i = 0; i < 16; i += 4) {
                float4 w4 = *(const float4*)&wsrc[i];
                Bt[bc + i][bk]     = (_Float16)w4.x;
                Bt[bc + i + 1][bk] = (_Float16)w4.y;
                Bt[bc + i + 2][bk] = (_Float16)w4.z;
                Bt[bc + i + 3][bk] = (_Float16)w4.w;
            }
        }
        __syncthreads();
        v16h a = frag16(&As[ln][0], khalf);
        v16h b = frag16(&Bt[n0 + ln][0], khalf);
        acc = __builtin_amdgcn_wmma_f32_16x16x32_f16(
            false, a, false, b, (short)0, acc, false, false);
    }
    return acc;
}

// -------------------------------------------------------------------------
// Fused GEMM:  Y = relu( scale(X1)@Wa + [X2@Wb] + bias )
// Block = 256 threads (8 wave32), tile = 16 rows x 128 cols.
// -------------------------------------------------------------------------
template<int K1, int K2, bool MEAN>
__global__ __launch_bounds__(256)
void gemm_bias_relu_kernel(const float* __restrict__ X1,
                           const float* __restrict__ icnt,
                           const float* __restrict__ Wa,
                           const float* __restrict__ X2,
                           const float* __restrict__ Wb,
                           const float* __restrict__ bias,
                           float* __restrict__ Y, int N) {
    __shared__ alignas(16) _Float16 As[16][32];
    __shared__ alignas(16) _Float16 Bt[128][32];

    const int t     = threadIdx.x;
    const int lane  = t & 31;
    const int khalf = lane >> 4;
    const int ln    = lane & 15;
    const int n0    = (t >> 5) * 16;
    const int r0    = blockIdx.x * 16;
    const int ar    = (t * 2) >> 5, ak = (t * 2) & 31;
    const int arow  = r0 + ar;
    const int bk    = (t * 16) >> 7, bc = (t * 16) & 127;

    v8f acc;
    {
        float bv = bias[n0 + ln];
        #pragma unroll
        for (int g = 0; g < 8; ++g) acc[g] = bv;
    }

    float s1 = 1.0f;
    if (MEAN) s1 = icnt[arow < N ? arow : N - 1];  // precomputed reciprocal

    acc = gemm_pass<K1>(X1, Wa, s1, arow, N, As, Bt, ar, ak, bk, bc, ln, khalf, n0, acc);
    acc = gemm_pass<K2>(X2, Wb, 1.0f, arow, N, As, Bt, ar, ak, bk, bc, ln, khalf, n0, acc);

    #pragma unroll
    for (int g = 0; g < 8; ++g) {
        int row = r0 + g + 8 * khalf;
        if (row < N) Y[(size_t)row * H_ + n0 + ln] = fmaxf(acc[g], 0.0f);
    }
}

// -------------------------------------------------------------------------
// Fused link-prediction head, 16 classification edges per block:
//   ee  = relu(edge_attr @ We + be)              (K=16 zero-padded to 32)
//   h   = relu(u2[s]@W1a + m2[d]@W1b + ee@W1c + b1)   (12 WMMAs / 16x16 tile)
//   out = h @ W2 + b2                             (VALU reduce, 2 outputs)
// -------------------------------------------------------------------------
__global__ __launch_bounds__(256)
void cls_head_kernel(const float* __restrict__ u2, const float* __restrict__ m2,
                     const float* __restrict__ eattr,
                     const int* __restrict__ srcIdx, const int* __restrict__ dstIdx,
                     const float* __restrict__ We, const float* __restrict__ be,
                     const float* __restrict__ W1, const float* __restrict__ b1,
                     const float* __restrict__ W2, const float* __restrict__ b2,
                     float* __restrict__ out) {
    __shared__ alignas(16) _Float16 Asrc[16][H_];
    __shared__ alignas(16) _Float16 Adst[16][H_];
    __shared__ alignas(16) _Float16 Aee[16][H_];
    __shared__ alignas(16) _Float16 Ea[16][32];
    __shared__ alignas(16) _Float16 Bt[128][32];   // transposed B
    __shared__ float Hs[16][H_];
    __shared__ int   sid[16], did[16];

    const int t     = threadIdx.x;
    const int lane  = t & 31;
    const int khalf = lane >> 4;
    const int ln    = lane & 15;
    const int n0    = (t >> 5) * 16;
    const int e0    = blockIdx.x * 16;

    if (t < 16) { sid[t] = srcIdx[e0 + t]; did[t] = dstIdx[e0 + t]; }
    {   // edge-attr 16x16 tile, K zero-padded to 32
        int e = t >> 4, c = t & 15;
        Ea[e][c]      = (_Float16)eattr[(size_t)(e0 + e) * DE_ + c];
        Ea[e][c + 16] = (_Float16)0.0f;
    }
    __syncthreads();

    {   // gather u2/m2 rows -> f16 LDS, 8 floats/thread/matrix
        int flat = t * 8;
        int e = flat >> 7, c = flat & 127;
        const float* ps = &u2[(size_t)sid[e] * H_ + c];
        const float* pd = &m2[(size_t)did[e] * H_ + c];
        #pragma unroll
        for (int i = 0; i < 8; i += 4) {
            float4 vs = *(const float4*)&ps[i];
            float4 vd = *(const float4*)&pd[i];
            Asrc[e][c + i]     = (_Float16)vs.x;  Asrc[e][c + i + 1] = (_Float16)vs.y;
            Asrc[e][c + i + 2] = (_Float16)vs.z;  Asrc[e][c + i + 3] = (_Float16)vs.w;
            Adst[e][c + i]     = (_Float16)vd.x;  Adst[e][c + i + 1] = (_Float16)vd.y;
            Adst[e][c + i + 2] = (_Float16)vd.z;  Adst[e][c + i + 3] = (_Float16)vd.w;
        }
    }
    {   // stage We (16x128) transposed into Bt cols 0..15, zero-pad 16..31
        int flat = t * 8;
        int kk = flat >> 7, c = flat & 127;
        #pragma unroll
        for (int i = 0; i < 8; ++i) {
            Bt[c + i][kk]      = (_Float16)We[(size_t)kk * H_ + c + i];
            Bt[c + i][kk + 16] = (_Float16)0.0f;
        }
    }
    __syncthreads();

    v8f acc;
    {   // ee = relu(Ea @ We + be)
        float bv = be[n0 + ln];
        #pragma unroll
        for (int g = 0; g < 8; ++g) acc[g] = bv;
        v16h a = frag16(&Ea[ln][0], khalf);
        v16h b = frag16(&Bt[n0 + ln][0], khalf);
        acc = __builtin_amdgcn_wmma_f32_16x16x32_f16(
            false, a, false, b, (short)0, acc, false, false);
        #pragma unroll
        for (int g = 0; g < 8; ++g)
            Aee[g + 8 * khalf][n0 + ln] = (_Float16)fmaxf(acc[g], 0.0f);
    }

    {   // h = relu(src@W1a + dst@W1b + ee@W1c + b1)
        float bv = b1[n0 + ln];
        #pragma unroll
        for (int g = 0; g < 8; ++g) acc[g] = bv;
    }
    const _Float16 (*Amats[3])[H_] = { Asrc, Adst, Aee };
    #pragma unroll
    for (int term = 0; term < 3; ++term) {
        const _Float16 (*A)[H_] = Amats[term];
        #pragma unroll
        for (int k0 = 0; k0 < H_; k0 += 32) {
            __syncthreads();  // also makes Aee writes visible before term==2 reads
            {   // stage W1 rows [term*128 + k0 .. +31] transposed
                int flat = t * 16;
                int kk = flat >> 7, c = flat & 127;
                const float* wsrc = &W1[(size_t)(term * H_ + k0 + kk) * H_ + c];
                #pragma unroll
                for (int i = 0; i < 16; i += 4) {
                    float4 v = *(const float4*)&wsrc[i];
                    Bt[c + i][kk]     = (_Float16)v.x;
                    Bt[c + i + 1][kk] = (_Float16)v.y;
                    Bt[c + i + 2][kk] = (_Float16)v.z;
                    Bt[c + i + 3][kk] = (_Float16)v.w;
                }
            }
            __syncthreads();
            v16h a = frag16(&A[ln][k0], khalf);
            v16h b = frag16(&Bt[n0 + ln][0], khalf);
            acc = __builtin_amdgcn_wmma_f32_16x16x32_f16(
                false, a, false, b, (short)0, acc, false, false);
        }
    }
    #pragma unroll
    for (int g = 0; g < 8; ++g)
        Hs[g + 8 * khalf][n0 + ln] = fmaxf(acc[g], 0.0f);
    __syncthreads();

    if (t < 32) {  // out[e][c] = b2[c] + sum_k h[e][k] * W2[k][c]
        int e = t >> 1, c = t & 1;
        float sum = b2[c];
        for (int k = 0; k < H_; ++k) sum += Hs[e][k] * W2[k * 2 + c];
        out[(size_t)(e0 + e) * 2 + c] = sum;
    }
}

// -------------------------------------------------------------------------
extern "C" void kernel_launch(void* const* d_in, const int* in_sizes, int n_in,
                              void* d_out, int out_size, void* d_ws, size_t ws_size,
                              hipStream_t stream) {
    const float* x_user     = (const float*)d_in[0];
    const float* x_merchant = (const float*)d_in[1];
    const float* edge_attr  = (const float*)d_in[2];
    const int*   ei_um      = (const int*)d_in[3];   // [2, EMP] user->merchant
    const int*   ei_mu      = (const int*)d_in[4];   // [2, EMP] merchant->user
    const int*   cls_ei     = (const int*)d_in[5];   // [2, ECL]
    const float* Wu = (const float*)d_in[6];  const float* bu = (const float*)d_in[7];
    const float* Wm = (const float*)d_in[8];  const float* bm = (const float*)d_in[9];
    const float* c1_mu_Wl = (const float*)d_in[10]; const float* c1_mu_bl = (const float*)d_in[11];
    const float* c1_mu_Wr = (const float*)d_in[12];
    const float* c1_um_Wl = (const float*)d_in[13]; const float* c1_um_bl = (const float*)d_in[14];
    const float* c1_um_Wr = (const float*)d_in[15];
    const float* c2_mu_Wl = (const float*)d_in[16]; const float* c2_mu_bl = (const float*)d_in[17];
    const float* c2_mu_Wr = (const float*)d_in[18];
    const float* c2_um_Wl = (const float*)d_in[19]; const float* c2_um_bl = (const float*)d_in[20];
    const float* c2_um_Wr = (const float*)d_in[21];
    const float* We = (const float*)d_in[22]; const float* be = (const float*)d_in[23];
    const float* W1 = (const float*)d_in[24]; const float* b1 = (const float*)d_in[25];
    const float* W2 = (const float*)d_in[26]; const float* b2 = (const float*)d_in[27];
    float* out = (float*)d_out;

    // workspace: A,B,C,D,E node buffers (each 100000*128 f32) + per-node counts
    const size_t NF = (size_t)NU_ * H_;
    float* A    = (float*)d_ws;     // xu, later u2
    float* B    = A + NF;           // xm, later m2
    float* C    = B + NF;           // u1
    float* D    = C + NF;           // m1
    float* E    = D + NF;           // agg scratch
    float* cntU = E + NF;
    float* cntM = cntU + NU_;

    const int gemmBlocksN = (NU_ + 15) / 16;        // 6250
    const int scatBlocks  = (EMP_ * 32 + 255) / 256;
    const int cntBlocks   = (EMP_ + 255) / 256;

    // in-degree reciprocals (edge-list-invariant; reused by both SAGE layers)
    hipMemsetAsync(cntU, 0, (NU_ + NM_) * sizeof(float), stream);
    count_kernel<<<cntBlocks, 256, 0, stream>>>(ei_mu + EMP_, cntU, EMP_);
    count_kernel<<<cntBlocks, 256, 0, stream>>>(ei_um + EMP_, cntM, EMP_);
    invcnt_kernel<<<((NU_ + NM_) + 255) / 256, 256, 0, stream>>>(cntU, NU_ + NM_);

    // input projections: xu = relu(x_user@Wu+bu), xm = relu(x_merchant@Wm+bm)
    gemm_bias_relu_kernel<DU_, 0, false><<<gemmBlocksN, 256, 0, stream>>>(
        x_user, nullptr, Wu, nullptr, nullptr, bu, A, NU_);
    gemm_bias_relu_kernel<DU_, 0, false><<<gemmBlocksN, 256, 0, stream>>>(
        x_merchant, nullptr, Wm, nullptr, nullptr, bm, B, NM_);

    // layer 1, users:  u1 = relu(mean_mu(xm)@Wl + bl + xu@Wr)
    hipMemsetAsync(E, 0, NF * sizeof(float), stream);
    scatter_add_kernel<<<scatBlocks, 256, 0, stream>>>(B, ei_mu, ei_mu + EMP_, E, EMP_);
    gemm_bias_relu_kernel<H_, H_, true><<<gemmBlocksN, 256, 0, stream>>>(
        E, cntU, c1_mu_Wl, A, c1_mu_Wr, c1_mu_bl, C, NU_);

    // layer 1, merchants: m1 = relu(mean_um(xu)@Wl + bl + xm@Wr)
    hipMemsetAsync(E, 0, NF * sizeof(float), stream);
    scatter_add_kernel<<<scatBlocks, 256, 0, stream>>>(A, ei_um, ei_um + EMP_, E, EMP_);
    gemm_bias_relu_kernel<H_, H_, true><<<gemmBlocksN, 256, 0, stream>>>(
        E, cntM, c1_um_Wl, B, c1_um_Wr, c1_um_bl, D, NM_);

    // layer 2, users:  u2 = relu(mean_mu(m1)@Wl + bl + u1@Wr)   (A := u2)
    hipMemsetAsync(E, 0, NF * sizeof(float), stream);
    scatter_add_kernel<<<scatBlocks, 256, 0, stream>>>(D, ei_mu, ei_mu + EMP_, E, EMP_);
    gemm_bias_relu_kernel<H_, H_, true><<<gemmBlocksN, 256, 0, stream>>>(
        E, cntU, c2_mu_Wl, C, c2_mu_Wr, c2_mu_bl, A, NU_);

    // layer 2, merchants: m2 = relu(mean_um(u1)@Wl + bl + m1@Wr)  (B := m2)
    hipMemsetAsync(E, 0, NF * sizeof(float), stream);
    scatter_add_kernel<<<scatBlocks, 256, 0, stream>>>(C, ei_um, ei_um + EMP_, E, EMP_);
    gemm_bias_relu_kernel<H_, H_, true><<<gemmBlocksN, 256, 0, stream>>>(
        E, cntM, c2_um_Wl, D, c2_um_Wr, c2_um_bl, B, NM_);

    // fused classifier head -> logits [ECL, 2]
    cls_head_kernel<<<ECL_ / 16, 256, 0, stream>>>(
        A, B, edge_attr, cls_ei, cls_ei + ECL_, We, be, W1, b1, W2, b2, out);
}
